// Encoder_82695300317581
// MI455X (gfx1250) — matrix operand order repaired and checked by reading, not compiled, not run on hardware
//
#include <hip/hip_runtime.h>

typedef __bf16 bf16;
typedef __attribute__((ext_vector_type(16))) bf16  v16bf;
typedef __attribute__((ext_vector_type(8)))  float v8f;

__device__ __forceinline__ float leakyf(float v){ return v > 0.f ? v : 0.01f * v; }

__device__ __forceinline__ v8f v8zero(){
    v8f z;
#pragma unroll
    for (int i = 0; i < 8; i++) z[i] = 0.f;
    return z;
}

// ---------------------------------------------------------------------------
// Weight prepack: fp32 [O, I, E, KH, KW] (E==1 for plain convs) -> bf16 A-tiles
// laid out as [e][kt][mt][lane 0..31][16 halves], matching the CDNA5 16-bit
// A-matrix 16x32 per-lane layout (lane m: K=0..7,16..23 ; lane m+16: K=8..15,24..31).
// K order: k = i*KH*KW + ky*KW + kx.
// ---------------------------------------------------------------------------
__global__ void pack_weights(const float* __restrict__ w, bf16* __restrict__ out,
                             int O, int I, int E, int KH, int KW, int KT, int MT)
{
    int gid = blockIdx.x * blockDim.x + threadIdx.x;
    int total = E * KT * MT * 32;
    if (gid >= total) return;
    int l    = gid & 31;
    int rest = gid >> 5;
    int mt = rest % MT; rest /= MT;
    int kt = rest % KT; rest /= KT;
    int e  = rest;
    int hi = l >> 4;
    int m  = l & 15;
    int o  = mt * 16 + m;
    int KK = KH * KW;
    int KREAL = I * KK;
    bf16* dst = out + (size_t)gid * 16;
#pragma unroll 4
    for (int j = 0; j < 16; j++) {
        int kl = (j < 8) ? (j + 8 * hi) : (j + 8 + 8 * hi);
        int k  = kt * 32 + kl;
        float v = 0.f;
        if (k < KREAL && o < O) {
            int i  = k / KK;
            int r  = k - i * KK;
            int ky = r / KW;
            int kx = r - ky * KW;
            v = w[((size_t)(o * I + i) * E + e) * KK + ky * KW + kx];
        }
        dst[j] = (bf16)v;
    }
}

// ---------------------------------------------------------------------------
// Stem conv: 3->64, k5, stride2, pad2, fused leaky-ReLU, fp32 in -> bf16 out.
// ---------------------------------------------------------------------------
__global__ void conv1_kernel(const float* __restrict__ x, const float* __restrict__ w,
                             const float* __restrict__ b, bf16* __restrict__ out)
{
    int gid = blockIdx.x * 256 + threadIdx.x;   // 8*64*128*128 total, exact
    int wc = gid & 127; int rest = gid >> 7;
    int hr = rest & 127; rest >>= 7;
    int o  = rest & 63;  int n = rest >> 6;
    float acc = b[o];
#pragma unroll
    for (int i = 0; i < 3; i++) {
#pragma unroll
        for (int ky = 0; ky < 5; ky++) {
            int y = 2 * hr + ky - 2;
            if ((unsigned)y >= 256u) continue;
#pragma unroll
            for (int kx = 0; kx < 5; kx++) {
                int xx = 2 * wc + kx - 2;
                if ((unsigned)xx >= 256u) continue;
                acc += x[((size_t)(n * 3 + i) * 256 + y) * 256 + xx] *
                       w[((o * 3 + i) * 5 + ky) * 5 + kx];
            }
        }
    }
    out[gid] = (bf16)leakyf(acc);
}

// ---------------------------------------------------------------------------
// Coupler: 1x1 conv (C -> 8 logits) at output stride + softmax + top-1.
// ---------------------------------------------------------------------------
template<int C>
__global__ void coupler_kernel(const bf16* __restrict__ in, int Hin, int Win, int stride,
                               const float* __restrict__ cw, const float* __restrict__ cb,
                               int* __restrict__ selIdx, float* __restrict__ selProb,
                               int Hout, int Wout, int total)
{
    int gid = blockIdx.x * blockDim.x + threadIdx.x;
    if (gid >= total) return;
    int wc = gid % Wout; int rest = gid / Wout;
    int hr = rest % Hout; int n = rest / Hout;
    int y = hr * stride, x = wc * stride;
    float lg[8];
#pragma unroll
    for (int e = 0; e < 8; e++) lg[e] = cb[e];
    for (int c = 0; c < C; c++) {
        float v = (float)in[((size_t)(n * C + c) * Hin + y) * Win + x];
#pragma unroll
        for (int e = 0; e < 8; e++) lg[e] += v * cw[e * C + c];
    }
    float m = lg[0]; int am = 0;
#pragma unroll
    for (int e = 1; e < 8; e++) if (lg[e] > m) { m = lg[e]; am = e; }
    float s = 0.f;
#pragma unroll
    for (int e = 0; e < 8; e++) s += __expf(lg[e] - m);
    selIdx[gid]  = am;
    selProb[gid] = 1.f / s;     // exp(max-max)/sum
}

// ---------------------------------------------------------------------------
// Implicit-GEMM conv via bf16 WMMA, f32 accumulate.
//   M = MTILES*16 output channels, N = 16 pixels (one output row segment),
//   K = CIN*KS*KS.
// The B matrix is pre-swizzled ONCE per tile into LDS in the exact WMMA
// per-lane layout (bmat[kt*32 + lane][16 halves]); the expert loop then reads
// B with a single contiguous 32B LDS load per lane per K-step, amortizing the
// gather across all experts present in the tile.
// MODE: 0 leaky->bf16 | 1 f32 + relu->bf16 | 2 relu->bf16
//       3 +res, f32 + relu->bf16 | 4 +res, leaky->f32 (final output)
// ---------------------------------------------------------------------------
template<int CIN, int KS, int STRIDE, int MTILES, int NWAVES, int NEXP, int MODE>
__global__ __launch_bounds__(NWAVES * 32)
void sconv_kernel(const bf16* __restrict__ in, int Hin, int Win,
                  const bf16* __restrict__ wa,
                  const float* __restrict__ bias,
                  const int* __restrict__ selIdx, const float* __restrict__ selProb,
                  const float* __restrict__ resIn,
                  float* __restrict__ outF, bf16* __restrict__ outB,
                  int Hout, int Wout)
{
    constexpr int KK    = KS * KS;
    constexpr int KREAL = CIN * KK;
    constexpr int KT    = (KREAL + 31) / 32;
    constexpr int PW    = STRIDE * 15 + KS;     // input cols needed by 16 pixels
    constexpr int PWP   = PW + 1;               // pad to dodge bank conflicts
    constexpr int MPW   = MTILES / NWAVES;
    constexpr int PAD   = KS / 2;
    constexpr int NT    = NWAVES * 32;
    constexpr int OCH   = MTILES * 16;

    __shared__ bf16 patch[CIN * KS * PWP];
    __shared__ bf16 bmat[KT * 32 * 16];         // WMMA-B-layout staging

    const int n   = blockIdx.y;
    const int WT  = Wout >> 4;
    const int h   = blockIdx.x / WT;
    const int w0  = (blockIdx.x % WT) << 4;
    const int tid = threadIdx.x;

    // --- Stage the input patch (zero padded) into LDS ---
    const int y0 = STRIDE * h  - PAD;
    const int x0 = STRIDE * w0 - PAD;
    for (int t = tid; t < CIN * KS * PW; t += NT) {
        int i  = t / (KS * PW);
        int r  = t - i * (KS * PW);
        int ky = r / PW;
        int px = r - ky * PW;
        int y  = y0 + ky;
        int x  = x0 + px;
        bf16 v = (bf16)0.f;
        if ((unsigned)y < (unsigned)Hin && (unsigned)x < (unsigned)Win)
            v = in[(((size_t)n * CIN + i) * Hin + y) * Win + x];
        patch[(i * KS + ky) * PWP + px] = v;
    }
    __syncthreads();

    // --- One-time B swizzle into WMMA per-lane layout ---
    // element (k, p):  kl = k&31, hi = (kl>>3)&1, j = (kl&7) + 8*(kl>>4), lane = p + 16*hi
    // bmat[( (k>>5)*32 + 16*hi + p )*16 + j] = patch[off(k) + STRIDE*p]
    for (int k = tid; k < KT * 32; k += NT) {
        int off = 0;
        bool valid = (k < KREAL);
        if (valid) {
            int i  = k / KK;
            int r  = k - i * KK;
            int ky = r / KS;
            int kx = r - ky * KS;
            off = (i * KS + ky) * PWP + kx;
        }
        int kl  = k & 31;
        int hi2 = (kl >> 3) & 1;
        int j   = (kl & 7) + ((kl >> 4) << 3);
        bf16* dst = &bmat[(((k >> 5) * 32 + 16 * hi2) << 4) + j];
#pragma unroll
        for (int p2 = 0; p2 < 16; p2++) {
            bf16 v = valid ? patch[off + STRIDE * p2] : (bf16)0.f;
            dst[p2 * 16] = v;
        }
    }
    __syncthreads();

    const int l  = tid & 31;
    const int wv = tid >> 5;
    const int p  = l & 15;
    const int hi = l >> 4;

    int se = 0; float pr = 1.f;
    if (NEXP > 1) {
        int pix = ((n * Hout) + h) * Wout + w0 + p;
        se = selIdx[pix];
        pr = selProb[pix];
    }

    v8f facc[MPW];
#pragma unroll
    for (int t = 0; t < MPW; t++) facc[t] = v8zero();

    const bf16* brow = bmat + (size_t)l * 16;   // per-lane B row base

    for (int e = 0; e < NEXP; e++) {
        if (NEXP > 1) {
            if (__ballot(se == e) == 0ull) continue;   // no pixel routed to e
        }
        v8f c[MPW];
#pragma unroll
        for (int t = 0; t < MPW; t++) c[t] = v8zero();

        const bf16* wexp = wa + (size_t)e * KT * MTILES * 512 +
                           (size_t)(wv * MPW) * 512 + l * 16;
        for (int kt = 0; kt < KT; kt++) {
            v16bf bm = *(const v16bf*)(brow + (size_t)kt * 512);
            const bf16* wk = wexp + (size_t)kt * MTILES * 512;
            __builtin_prefetch(wk + MTILES * 512, 0, 1);   // next K-step A block
#pragma unroll
            for (int t = 0; t < MPW; t++) {
                v16bf am = *(const v16bf*)(wk + (size_t)t * 512);
                c[t] = __builtin_amdgcn_wmma_f32_16x16x32_bf16(
                           false, am, false, bm, (short)0, c[t], false, false);
            }
        }

        float sc = (NEXP > 1) ? ((se == e) ? pr : 0.f) : 1.f;
#pragma unroll
        for (int t = 0; t < MPW; t++)
#pragma unroll
            for (int q = 0; q < 8; q++)
                facc[t][q] += c[t][q] * sc;
    }

    // Epilogue: bias + activation / residual; D layout: M = r + 8*hi, N = lane&15
#pragma unroll
    for (int t = 0; t < MPW; t++) {
        const int mt = wv * MPW + t;
#pragma unroll
        for (int r = 0; r < 8; r++) {
            const int o = mt * 16 + r + 8 * hi;
            float v = facc[t][r] + bias[o];
            const size_t oidx = (((size_t)n * OCH + o) * Hout + h) * Wout + w0 + p;
            if (MODE == 0) {
                outB[oidx] = (bf16)leakyf(v);
            } else if (MODE == 1) {
                outF[oidx] = v;
                outB[oidx] = (bf16)fmaxf(v, 0.f);
            } else if (MODE == 2) {
                outB[oidx] = (bf16)fmaxf(v, 0.f);
            } else if (MODE == 3) {
                v += resIn[oidx];
                outF[oidx] = v;
                outB[oidx] = (bf16)fmaxf(v, 0.f);
            } else {
                v += resIn[oidx];
                outF[oidx] = leakyf(v);
            }
        }
    }
}

// ---------------------------------------------------------------------------
extern "C" void kernel_launch(void* const* d_in, const int* in_sizes, int n_in,
                              void* d_out, int out_size, void* d_ws, size_t ws_size,
                              hipStream_t stream)
{
    (void)in_sizes; (void)n_in; (void)out_size; (void)ws_size;

    const float* x    = (const float*)d_in[0];
    const float* w1   = (const float*)d_in[1];
    const float* b1   = (const float*)d_in[2];
    const float* c1w  = (const float*)d_in[3];
    const float* c1b  = (const float*)d_in[4];
    const float* s1w  = (const float*)d_in[5];
    const float* s1b  = (const float*)d_in[6];
    const float* c2w  = (const float*)d_in[7];
    const float* c2b  = (const float*)d_in[8];
    const float* s2w  = (const float*)d_in[9];
    const float* s2b  = (const float*)d_in[10];
    const float* r0w1 = (const float*)d_in[11];
    const float* r0b1 = (const float*)d_in[12];
    const float* r0w2 = (const float*)d_in[13];
    const float* r0b2 = (const float*)d_in[14];
    const float* r1w1 = (const float*)d_in[15];
    const float* r1b1 = (const float*)d_in[16];
    const float* r1w2 = (const float*)d_in[17];
    const float* r1b2 = (const float*)d_in[18];

    char* ws = (char*)d_ws;
    size_t off = 0;
    auto alloc = [&](size_t bytes) {
        char* pp = ws + off;
        off += (bytes + 255) & ~(size_t)255;
        return pp;
    };
    bf16*  h1    = (bf16*) alloc(8ull * 64 * 128 * 128 * 2);   // stem out (leaky, bf16)
    bf16*  h2    = (bf16*) alloc(8ull * 128 * 64 * 64 * 2);    // sconv1 out (leaky, bf16)
    float* xf    = (float*)alloc(8ull * 128 * 64 * 64 * 4);    // residual trunk, fp32
    bf16*  tact  = (bf16*) alloc(8ull * 128 * 64 * 64 * 2);    // relu(trunk), bf16
    bf16*  tmid  = (bf16*) alloc(8ull * 32  * 64 * 64 * 2);    // res bottleneck, bf16
    int*   s1i   = (int*)  alloc(8ull * 64 * 64 * 4);
    float* s1p   = (float*)alloc(8ull * 64 * 64 * 4);
    int*   s2i   = (int*)  alloc(8ull * 64 * 64 * 4);
    float* s2p   = (float*)alloc(8ull * 64 * 64 * 4);
    bf16*  WA1   = (bf16*) alloc(8ull * 50 * 8 * 512 * 2);     // s1w swizzled
    bf16*  WA2   = (bf16*) alloc(8ull * 36 * 8 * 512 * 2);     // s2w swizzled
    bf16*  WR1a  = (bf16*) alloc(36ull * 2 * 512 * 2);         // r0w1
    bf16*  WR1b  = (bf16*) alloc(36ull * 2 * 512 * 2);         // r1w1
    bf16*  WR2a  = (bf16*) alloc(1ull  * 8 * 512 * 2);         // r0w2
    bf16*  WR2b  = (bf16*) alloc(1ull  * 8 * 512 * 2);         // r1w2

    // --- weight prepack ---
    pack_weights<<<(8 * 50 * 8 * 32 + 255) / 256, 256, 0, stream>>>(s1w, WA1, 128, 64, 8, 5, 5, 50, 8);
    pack_weights<<<(8 * 36 * 8 * 32 + 255) / 256, 256, 0, stream>>>(s2w, WA2, 128, 128, 8, 3, 3, 36, 8);
    pack_weights<<<(36 * 2 * 32 + 255) / 256, 256, 0, stream>>>(r0w1, WR1a, 32, 128, 1, 3, 3, 36, 2);
    pack_weights<<<(36 * 2 * 32 + 255) / 256, 256, 0, stream>>>(r1w1, WR1b, 32, 128, 1, 3, 3, 36, 2);
    pack_weights<<<(8 * 32 + 255) / 256, 256, 0, stream>>>(r0w2, WR2a, 128, 32, 1, 1, 1, 1, 8);
    pack_weights<<<(8 * 32 + 255) / 256, 256, 0, stream>>>(r1w2, WR2b, 128, 32, 1, 1, 1, 1, 8);

    // --- stem conv + leaky ---
    conv1_kernel<<<(8 * 64 * 128 * 128) / 256, 256, 0, stream>>>(x, w1, b1, h1);

    // --- switched conv 1: route + WMMA conv (64->128, k5, s2) + leaky ---
    coupler_kernel<64><<<(8 * 64 * 64 + 255) / 256, 256, 0, stream>>>(
        h1, 128, 128, 2, c1w, c1b, s1i, s1p, 64, 64, 8 * 64 * 64);
    sconv_kernel<64, 5, 2, 8, 4, 8, 0><<<dim3(64 * 4, 8), 128, 0, stream>>>(
        h1, 128, 128, WA1, s1b, s1i, s1p, nullptr, nullptr, h2, 64, 64);

    // --- switched conv 2: route + WMMA conv (128->128, k3, s1), no act ---
    coupler_kernel<128><<<(8 * 64 * 64 + 255) / 256, 256, 0, stream>>>(
        h2, 64, 64, 1, c2w, c2b, s2i, s2p, 64, 64, 8 * 64 * 64);
    sconv_kernel<128, 3, 1, 8, 4, 8, 1><<<dim3(64 * 4, 8), 128, 0, stream>>>(
        h2, 64, 64, WA2, s2b, s2i, s2p, nullptr, xf, tact, 64, 64);

    // --- res block 0: conv3x3(relu(x)) -> relu -> conv1x1 + residual ---
    sconv_kernel<128, 3, 1, 2, 2, 1, 2><<<dim3(64 * 4, 8), 64, 0, stream>>>(
        tact, 64, 64, WR1a, r0b1, nullptr, nullptr, nullptr, nullptr, tmid, 64, 64);
    sconv_kernel<32, 1, 1, 8, 4, 1, 3><<<dim3(64 * 4, 8), 128, 0, stream>>>(
        tmid, 64, 64, WR2a, r0b2, nullptr, nullptr, xf, xf, tact, 64, 64);

    // --- res block 1 + final leaky -> d_out (fp32) ---
    sconv_kernel<128, 3, 1, 2, 2, 1, 2><<<dim3(64 * 4, 8), 64, 0, stream>>>(
        tact, 64, 64, WR1b, r1b1, nullptr, nullptr, nullptr, nullptr, tmid, 64, 64);
    sconv_kernel<32, 1, 1, 8, 4, 1, 4><<<dim3(64 * 4, 8), 128, 0, stream>>>(
        tmid, 64, 64, WR2b, r1b2, nullptr, nullptr, xf, (float*)d_out, nullptr, 64, 64);
}
